// GeneGNN_44375602102303
// MI455X (gfx1250) — compile-verified, matrix operand order checked by reading, not run on hardware
//
#include <hip/hip_runtime.h>
#include <hip/hip_bf16.h>
#include <math.h>

#define GENE   3008
#define NLEAF  512
#define HH     6
#define GPL    16
#define BB     64
#define NL1    64
#define NL2    8
#define EPSV   1e-5f

typedef __attribute__((ext_vector_type(16))) __bf16 v16bf;
typedef __attribute__((ext_vector_type(8)))  float  v8f;

// ---------------------------------------------------------------------------
// Kernel 1: per-leaf fused GEMM (64x3008 @ 3008x16, bf16 WMMA, f32 accum)
//           + bias + (16x6 leaf MLP) + tanh + BatchNorm over B=64.
// grid = 512 (one WG per leaf), block = 128 threads = 4 waves (4 M-tiles).
// ---------------------------------------------------------------------------
__global__ __launch_bounds__(128) void gene_leaf_kernel(
    const float* __restrict__ x,    // (64, 3008)
    const float* __restrict__ Wdg,  // (512, 3008, 16)
    const float* __restrict__ bdg,  // (512, 16)
    const float* __restrict__ W0,   // (512, 16, 6)
    const float* __restrict__ b0,   // (512, 6)
    const float* __restrict__ g0,   // (512, 6)
    const float* __restrict__ be0,  // (512, 6)
    float* __restrict__ h0out)      // (64, 512, 6)
{
    const int t    = blockIdx.x;
    const int lane = threadIdx.x & 31;
    const int wave = threadIdx.x >> 5;   // M-tile: rows wave*16 .. wave*16+15
    const int half = lane >> 4;          // K-group selector (wave32 halves)
    const int l16  = lane & 15;

    const float* __restrict__ xrow = x + (size_t)(wave * 16 + l16) * GENE;
    const float* __restrict__ Wt   = Wdg + (size_t)t * GENE * GPL;

    v8f c = {};  // f32 accumulator, 16x16 tile

    for (int k0 = 0; k0 < GENE; k0 += 32) {
        if (k0 + 32 < GENE) {
            // speculative prefetch of next W_dg chunk -> global_prefetch_b8
            __builtin_prefetch(Wt + (size_t)(k0 + 32) * GPL, 0, 1);
        }
        const int ka = k0 + half * 8;

        // A fragment: row M = l16 of this M-tile; 8 consecutive K (32B, b128x2)
        v16bf a;
        #pragma unroll
        for (int i = 0; i < 8; ++i) {
            a[i]     = (__bf16)xrow[ka + i];
            a[i + 8] = (__bf16)xrow[ka + 16 + i];
        }
        // B fragment: column N = l16 of W_dg[t]; rows k coalesce across lanes
        v16bf b;
        #pragma unroll
        for (int i = 0; i < 8; ++i) {
            b[i]     = (__bf16)Wt[(size_t)(ka + i) * GPL + l16];
            b[i + 8] = (__bf16)Wt[(size_t)(ka + 16 + i) * GPL + l16];
        }
        c = __builtin_amdgcn_wmma_f32_16x16x32_bf16(
                /*neg_a=*/false, a, /*neg_b=*/false, b,
                /*c_mod=*/(short)0, c, /*reuse_a=*/false, /*reuse_b=*/false);
    }

    // Spill gout tile (+bias) to LDS: C VGPR r -> (M = 16*wave + 8*half + r, N = l16)
    __shared__ float sG[BB][GPL];
    const float bias = bdg[t * GPL + l16];
    #pragma unroll
    for (int r = 0; r < 8; ++r)
        sG[wave * 16 + 8 * half + r][l16] = c[r] + bias;
    __syncthreads();

    // Leaf MLP: h = tanh(gout @ W0[t] + b0[t]); 64*6 outputs over 128 threads
    __shared__ float sH[BB][HH];
    for (int i = threadIdx.x; i < BB * HH; i += 128) {
        const int row = i / HH, h = i % HH;
        float acc = b0[t * HH + h];
        #pragma unroll
        for (int k = 0; k < GPL; ++k)
            acc += sG[row][k] * W0[(t * GPL + k) * HH + h];
        sH[row][h] = tanhf(acc);
    }
    __syncthreads();

    // BatchNorm over the 64-row batch (biased variance, eps = 1e-5)
    __shared__ float sMu[HH], sRs[HH];
    if (threadIdx.x < HH) {
        const int h = threadIdx.x;
        float mu = 0.f;
        for (int r = 0; r < BB; ++r) mu += sH[r][h];
        mu *= (1.f / BB);
        float var = 0.f;
        for (int r = 0; r < BB; ++r) { float d = sH[r][h] - mu; var += d * d; }
        var *= (1.f / BB);
        sMu[h] = mu;
        sRs[h] = rsqrtf(var + EPSV);
    }
    __syncthreads();

    for (int i = threadIdx.x; i < BB * HH; i += 128) {
        const int row = i / HH, h = i % HH;
        const float v = g0[t * HH + h] * (sH[row][h] - sMu[h]) * sRs[h] + be0[t * HH + h];
        h0out[(size_t)row * NLEAF * HH + t * HH + h] = v;
    }
}

// ---------------------------------------------------------------------------
// Kernel 2: level-1 nodes. grid = 64 (one WG per node), block = 64 (= batch).
// c1[b] = h0[b, j*8:(j+1)*8, :] (48 contiguous floats); y = tanh(c1@W1+b1); BN.
// ---------------------------------------------------------------------------
__global__ __launch_bounds__(64) void gene_l1_kernel(
    const float* __restrict__ h0,   // (64, 512, 6)
    const float* __restrict__ W1,   // (64, 48, 6)
    const float* __restrict__ b1, const float* __restrict__ g1,
    const float* __restrict__ be1,
    float* __restrict__ h1out)      // (64, 64, 6)
{
    const int j = blockIdx.x;
    const int b = threadIdx.x;
    const float* __restrict__ cvec = h0 + (size_t)b * NLEAF * HH + j * 8 * HH;

    float acc[HH];
    #pragma unroll
    for (int h = 0; h < HH; ++h) acc[h] = b1[j * HH + h];
    for (int k = 0; k < 48; ++k) {
        const float cv = cvec[k];
        #pragma unroll
        for (int h = 0; h < HH; ++h) acc[h] += cv * W1[(j * 48 + k) * HH + h];
    }

    __shared__ float sH[BB][HH];
    #pragma unroll
    for (int h = 0; h < HH; ++h) sH[b][h] = tanhf(acc[h]);
    __syncthreads();

    __shared__ float sMu[HH], sRs[HH];
    if (b < HH) {
        float mu = 0.f;
        for (int r = 0; r < BB; ++r) mu += sH[r][b];
        mu *= (1.f / BB);
        float var = 0.f;
        for (int r = 0; r < BB; ++r) { float d = sH[r][b] - mu; var += d * d; }
        var *= (1.f / BB);
        sMu[b] = mu;
        sRs[b] = rsqrtf(var + EPSV);
    }
    __syncthreads();

    #pragma unroll
    for (int h = 0; h < HH; ++h)
        h1out[(size_t)b * NL1 * HH + j * HH + h] =
            g1[j * HH + h] * (sH[b][h] - sMu[h]) * sRs[h] + be1[j * HH + h];
}

// ---------------------------------------------------------------------------
// Kernel 3: level-2 (8 nodes) + root, fused in one workgroup (tiny).
// grid = 1, block = 64 (= batch).
// ---------------------------------------------------------------------------
__global__ __launch_bounds__(64) void gene_root_kernel(
    const float* __restrict__ h1,   // (64, 64, 6)
    const float* __restrict__ W2,   // (8, 48, 6)
    const float* __restrict__ b2, const float* __restrict__ g2,
    const float* __restrict__ be2,
    const float* __restrict__ Wr,   // (48, 6)
    const float* __restrict__ br, const float* __restrict__ gr,
    const float* __restrict__ ber,
    float* __restrict__ out)        // (64, 6)
{
    const int b = threadIdx.x;

    __shared__ float sZ[BB][48];
    for (int j = 0; j < NL2; ++j) {
        const float* __restrict__ cvec = h1 + (size_t)b * NL1 * HH + j * 48;
        #pragma unroll
        for (int h = 0; h < HH; ++h) {
            float acc = b2[j * HH + h];
            for (int k = 0; k < 48; ++k) acc += cvec[k] * W2[(j * 48 + k) * HH + h];
            sZ[b][j * HH + h] = tanhf(acc);
        }
    }
    __syncthreads();

    __shared__ float sMu[48], sRs[48];
    if (b < 48) {
        float mu = 0.f;
        for (int r = 0; r < BB; ++r) mu += sZ[r][b];
        mu *= (1.f / BB);
        float var = 0.f;
        for (int r = 0; r < BB; ++r) { float d = sZ[r][b] - mu; var += d * d; }
        var *= (1.f / BB);
        sMu[b] = mu;
        sRs[b] = rsqrtf(var + EPSV);
    }
    __syncthreads();

    float cr[48];
    #pragma unroll
    for (int k = 0; k < 48; ++k)
        cr[k] = g2[k] * (sZ[b][k] - sMu[k]) * sRs[k] + be2[k];

    __shared__ float sO[BB][HH];
    #pragma unroll
    for (int h = 0; h < HH; ++h) {
        float acc = br[h];
        for (int k = 0; k < 48; ++k) acc += cr[k] * Wr[k * HH + h];
        sO[b][h] = tanhf(acc);
    }
    __syncthreads();

    __shared__ float sMu2[HH], sRs2[HH];
    if (b < HH) {
        float mu = 0.f;
        for (int r = 0; r < BB; ++r) mu += sO[r][b];
        mu *= (1.f / BB);
        float var = 0.f;
        for (int r = 0; r < BB; ++r) { float d = sO[r][b] - mu; var += d * d; }
        var *= (1.f / BB);
        sMu2[b] = mu;
        sRs2[b] = rsqrtf(var + EPSV);
    }
    __syncthreads();

    #pragma unroll
    for (int h = 0; h < HH; ++h)
        out[b * HH + h] = gr[h] * (sO[b][h] - sMu2[h]) * sRs2[h] + ber[h];
}

// ---------------------------------------------------------------------------
extern "C" void kernel_launch(void* const* d_in, const int* in_sizes, int n_in,
                              void* d_out, int out_size, void* d_ws, size_t ws_size,
                              hipStream_t stream) {
    const float* x   = (const float*)d_in[0];
    const float* Wdg = (const float*)d_in[1];
    const float* bdg = (const float*)d_in[2];
    const float* W0  = (const float*)d_in[3];
    const float* b0  = (const float*)d_in[4];
    const float* g0  = (const float*)d_in[5];
    const float* be0 = (const float*)d_in[6];
    const float* W1  = (const float*)d_in[7];
    const float* b1  = (const float*)d_in[8];
    const float* g1  = (const float*)d_in[9];
    const float* be1 = (const float*)d_in[10];
    const float* W2  = (const float*)d_in[11];
    const float* b2  = (const float*)d_in[12];
    const float* g2  = (const float*)d_in[13];
    const float* be2 = (const float*)d_in[14];
    const float* Wr  = (const float*)d_in[15];
    const float* br  = (const float*)d_in[16];
    const float* gr  = (const float*)d_in[17];
    const float* ber = (const float*)d_in[18];

    float* h0 = (float*)d_ws;                              // 64*512*6 floats
    float* h1 = h0 + (size_t)BB * NLEAF * HH;              // 64*64*6 floats
    float* outp = (float*)d_out;                           // (64, 6)

    gene_leaf_kernel<<<NLEAF, 128, 0, stream>>>(x, Wdg, bdg, W0, b0, g0, be0, h0);
    gene_l1_kernel<<<NL1, 64, 0, stream>>>(h0, W1, b1, g1, be1, h1);
    gene_root_kernel<<<1, 64, 0, stream>>>(h1, W2, b2, g2, be2, Wr, br, gr, ber, outp);
}
